// ACE_47949014892740
// MI455X (gfx1250) — matrix operand order repaired and checked by reading, not compiled, not run on hardware
//
#include <hip/hip_runtime.h>
#include <hip/hip_bf16.h>

#define B_ 8
#define C_ 256
#define H_ 128
#define W_ 128
#define J_ 19
#define S_ 512
#define NH_ 128
#define BJ_ (B_ * J_)      // 152
#define BJPAD 160
#define NTAB 4608          // 2 * 9 * 256  (gamma block then beta block)
#define KSP 1152           // 9 * 128

typedef __attribute__((ext_vector_type(16))) __bf16 v16bf;
typedef __attribute__((ext_vector_type(8)))  float  v8f;

struct U4 { unsigned int a, b, c, d; };

union V16BF {
  v16bf bf;
  unsigned short us[16];
  U4 q[2];
};

__device__ __forceinline__ unsigned short f32_to_bf16(float f) {
  unsigned int u = __float_as_uint(f);
  u += 0x7FFFu + ((u >> 16) & 1u);   // round-to-nearest-even
  return (unsigned short)(u >> 16);
}

// ---------------- 1) per-(b,c) instance-norm stats over x + noise ------------
__global__ void k_stats(const float* __restrict__ x, const float* __restrict__ noise,
                        const float* __restrict__ noise_var,
                        float* __restrict__ mean, float* __restrict__ rstd) {
  const int bc = blockIdx.x;
  const int b = bc / C_, c = bc % C_;
  const float nv = noise_var[c];
  const float* xp = x + (size_t)bc * (H_ * W_);
  const float* np = noise + (size_t)b * (W_ * H_);
  float s = 0.f, s2 = 0.f;
  for (int i = threadIdx.x; i < H_ * W_; i += 256) {
    int h = i / W_, w = i % W_;
    float v = xp[i] + np[w * H_ + h] * nv;
    s += v; s2 += v * v;
  }
  __shared__ float sm[256], sm2[256];
  sm[threadIdx.x] = s; sm2[threadIdx.x] = s2;
  __syncthreads();
  for (int st = 128; st > 0; st >>= 1) {
    if (threadIdx.x < st) { sm[threadIdx.x] += sm[threadIdx.x + st]; sm2[threadIdx.x] += sm2[threadIdx.x + st]; }
    __syncthreads();
  }
  if (threadIdx.x == 0) {
    float m = sm[0] * (1.f / (H_ * W_));
    float var = sm2[0] * (1.f / (H_ * W_)) - m * m;
    mean[bc] = m;
    rstd[bc] = rsqrtf(var + 1e-5f);
  }
}

// ---------------- 2) mu[b,j,o] = relu(style @ Wj^T + bj), output bf16 --------
__global__ void k_mu(const float* __restrict__ style, const float* __restrict__ fc_w,
                     const float* __restrict__ fc_b, unsigned short* __restrict__ mu_bf) {
  const int bj = blockIdx.x;
  if (bj >= BJ_) {  // zero padding rows 152..159 for the GEMM
    for (int o = threadIdx.x; o < S_; o += 256) mu_bf[(size_t)bj * S_ + o] = 0;
    return;
  }
  const int j = bj % J_;
  __shared__ float sc[S_];
  for (int i = threadIdx.x; i < S_; i += 256) sc[i] = style[(size_t)bj * S_ + i];
  __syncthreads();
  for (int o = threadIdx.x; o < S_; o += 256) {
    const float* wr = fc_w + ((size_t)j * S_ + o) * S_;
    float a = fc_b[j * S_ + o];
    for (int i = 0; i < S_; ++i) a = fmaf(sc[i], wr[i], a);
    mu_bf[(size_t)bj * S_ + o] = f32_to_bf16(fmaxf(a, 0.f));
  }
}

// ------ 3a) WtT[col][s]  col = which*2304 + tap*256 + c  (bf16, col-major K) -
__global__ void k_tr_gb(const float* __restrict__ gw, const float* __restrict__ bw,
                        unsigned short* __restrict__ WtT) {
  size_t i = (size_t)blockIdx.x * 256 + threadIdx.x;
  if (i >= (size_t)NTAB * S_) return;
  int col = (int)(i / S_), s = (int)(i % S_);
  int which = col >= 2304;
  int cc = which ? col - 2304 : col;
  int tap = cc / C_, c = cc % C_;
  const float* w = which ? bw : gw;
  WtT[i] = f32_to_bf16(w[((size_t)(c * S_ + s)) * 9 + tap]);
}

// ------ 3b) WspT[col][k]  col = 2*c + which, k = tap*128 + n  (bf16) ---------
__global__ void k_tr_sp(const float* __restrict__ gw, const float* __restrict__ bw,
                        unsigned short* __restrict__ WspT) {
  size_t i = (size_t)blockIdx.x * 256 + threadIdx.x;
  if (i >= (size_t)(2 * C_) * KSP) return;
  int col = (int)(i / KSP), k = (int)(i % KSP);
  int c = col >> 1, which = col & 1;
  int tap = k / NH_, n = k % NH_;
  const float* w = which ? bw : gw;
  WspT[i] = f32_to_bf16(w[((size_t)(c * NH_ + n)) * 9 + tap]);
}

// ------ 3c) Tsh[j][tap][n] = sp_shared_w[n,j,tap]  (f32) ---------------------
__global__ void k_tr_sh(const float* __restrict__ shw, float* __restrict__ Tsh) {
  int i = blockIdx.x * 256 + threadIdx.x;
  if (i >= J_ * 9 * NH_) return;
  int j = i / (9 * NH_), rem = i % (9 * NH_);
  int tap = rem / NH_, n = rem % NH_;
  Tsh[i] = shw[((size_t)(n * J_ + j)) * 9 + tap];
}

// ------ 4) Gtab[bj][col] = sum_s mu[bj,s] * WtT[col][s]   (WMMA bf16) --------
// M=160 (10 tiles), N=4608 (288 tiles), K=512.  4 waves/block, 1 tile/wave.
__global__ void k_gemm_tab(const unsigned short* __restrict__ mu_bf,
                           const unsigned short* __restrict__ WtT,
                           float* __restrict__ Gtab) {
  const int wid  = threadIdx.x >> 5;
  const int lane = threadIdx.x & 31;
  const int l15  = lane & 15;
  const int hi8  = (lane >> 4) << 3;   // 0 or 8
  const int hi16 = (lane >> 4) << 4;   // 0 or 16
  const int mtile = blockIdx.x;
  const int ntile = blockIdx.y * 4 + wid;
  const int col = ntile * 16 + l15;
  v8f acc;
  for (int i = 0; i < 8; ++i) acc[i] = 0.f;
  const unsigned short* pa0 = mu_bf + (size_t)(mtile * 16 + l15) * S_;
  const unsigned short* pb0 = WtT + (size_t)col * S_;
  for (int kb = 0; kb < S_; kb += 32) {
    V16BF A, Bm;
    const unsigned short* pa = pa0 + kb + hi8;   // A: K = e + (e>=8?8:0) + 8*hi
    A.q[0] = *(const U4*)(pa);
    A.q[1] = *(const U4*)(pa + 16);
    const unsigned short* pb = pb0 + kb + hi8;   // B: K = e + (e>=8?8:0) + 8*hi
    Bm.q[0] = *(const U4*)(pb);
    Bm.q[1] = *(const U4*)(pb + 16);
    acc = __builtin_amdgcn_wmma_f32_16x16x32_bf16(false, A.bf, false, Bm.bf,
                                                  (short)0, acc, false, false);
  }
  for (int r = 0; r < 8; ++r) {
    int m = mtile * 16 + r + hi8;
    if (m < BJ_) Gtab[(size_t)m * NTAB + col] = acc[r];
  }
}

// ------ 5) actv[b,h,w,n] = relu(shb[n] + sum_tap Tsh[label][tap][n]) (bf16) --
__global__ void k_actv(const int* __restrict__ seg, const float* __restrict__ Tsh,
                       const float* __restrict__ shb, unsigned short* __restrict__ actv) {
  const int bh = blockIdx.x;
  const int b = bh / H_, h = bh % H_;
  __shared__ int lab[3][W_ + 2];
  for (int i = threadIdx.x; i < 3 * (W_ + 2); i += 256) {
    int r = i / (W_ + 2), cc = i % (W_ + 2);
    int hh = h + r - 1, ww = cc - 1;
    lab[r][cc] = (hh >= 0 && hh < H_ && ww >= 0 && ww < W_) ? seg[(b * H_ + hh) * W_ + ww] : -1;
  }
  __syncthreads();
  for (int idx = threadIdx.x; idx < W_ * NH_; idx += 256) {
    int w = idx / NH_, n = idx % NH_;
    float a = shb[n];
    for (int tap = 0; tap < 9; ++tap) {
      int l = lab[tap / 3][w + tap % 3];
      if (l >= 0) a += Tsh[(l * 9 + tap) * NH_ + n];
    }
    actv[((size_t)bh * W_ + w) * NH_ + n] = f32_to_bf16(fmaxf(a, 0.f));
  }
}

// ------ 6) fused: spade implicit GEMM (WMMA) + gathers + norm + blend --------
// Block: 128 threads (4 waves). Tile: M=64 pixels (one (b,h) strip), N=128
// interleaved columns (col = 2c + {gamma,beta}); wave tile M=64 x N=32.
// Grid: (B*H*2, 4).  K = 1152 (tap-major, n within tap).
__global__ void k_main(const float* __restrict__ x, const int* __restrict__ seg,
                       const float* __restrict__ noise, const float* __restrict__ noise_var,
                       const float* __restrict__ bgam, const float* __restrict__ bbet,
                       const float* __restrict__ cgb, const float* __restrict__ cbb,
                       const float* __restrict__ sgb, const float* __restrict__ sbb,
                       const float* __restrict__ mean, const float* __restrict__ rstd,
                       const float* __restrict__ Gtab,
                       const unsigned short* __restrict__ actv,
                       const unsigned short* __restrict__ WspT,
                       float* __restrict__ out) {
  const int wid  = threadIdx.x >> 5;
  const int lane = threadIdx.x & 31;
  const int l15  = lane & 15;
  const int hi8  = (lane >> 4) << 3;
  const int hi16 = (lane >> 4) << 4;

  const int sid = blockIdx.x;
  const int b = sid / (H_ * 2);
  const int rem = sid % (H_ * 2);
  const int h = rem >> 1;
  const int wbase = (rem & 1) * 64;
  const int colbase = blockIdx.y * 128 + wid * 32;

  v8f acc[4][2];
  for (int mi = 0; mi < 4; ++mi)
    for (int ni = 0; ni < 2; ++ni)
      for (int i = 0; i < 8; ++i) acc[mi][ni][i] = 0.f;

  for (int kb = 0; kb < KSP; kb += 32) {
    const int tap = kb / NH_;            // constant within a 32-wide K chunk
    const int dy = tap / 3 - 1, dx = tap % 3 - 1;
    const int n0 = (kb % NH_) + hi8;
    const int hh = h + dy;
    V16BF A[4], Bm[2];
    for (int ni = 0; ni < 2; ++ni) {
      const unsigned short* pb =
          WspT + (size_t)(colbase + ni * 16 + l15) * KSP + kb + hi8;
      Bm[ni].q[0] = *(const U4*)(pb);
      Bm[ni].q[1] = *(const U4*)(pb + 16);
    }
    for (int mi = 0; mi < 4; ++mi) {
      const int ww = wbase + mi * 16 + l15 + dx;
      if (hh >= 0 && hh < H_ && ww >= 0 && ww < W_) {
        const unsigned short* pa =
            actv + ((size_t)((b * H_ + hh) * W_) + ww) * NH_ + n0;
        A[mi].q[0] = *(const U4*)(pa);
        A[mi].q[1] = *(const U4*)(pa + 16);
      } else {
        A[mi].q[0] = U4{0u, 0u, 0u, 0u};
        A[mi].q[1] = U4{0u, 0u, 0u, 0u};
      }
    }
    for (int mi = 0; mi < 4; ++mi)
      for (int ni = 0; ni < 2; ++ni)
        acc[mi][ni] = __builtin_amdgcn_wmma_f32_16x16x32_bf16(
            false, A[mi].bf, false, Bm[ni].bf, (short)0, acc[mi][ni], false, false);
  }

  // epilogue: pair gamma/beta via shfl_xor(1), fuse norm + blend, store f32.
  const float ga = 1.f / (1.f + __expf(-bgam[0]));
  const float ba = 1.f / (1.f + __expf(-bbet[0]));
  for (int ni = 0; ni < 2; ++ni) {
    const int col = colbase + ni * 16 + l15;
    const int c = col >> 1;
    const int isBeta = col & 1;
    const float gsbias = sgb[c], bsbias = sbb[c];
    const float cg = cgb[c], cb = cbb[c];
    const float mn = mean[b * C_ + c], rs = rstd[b * C_ + c];
    const float nv = noise_var[c];
    for (int mi = 0; mi < 4; ++mi) {
      for (int r = 0; r < 8; ++r) {
        float own = acc[mi][ni][r];
        float other = __shfl_xor(own, 1, 32);   // partner holds the other of {gamma,beta}
        bool doStore = isBeta ? (r >= 4) : (r < 4);
        if (!doStore) continue;
        float gsp = (isBeta ? other : own) + gsbias;
        float bsp = (isBeta ? own : other) + bsbias;
        const int w = wbase + mi * 16 + r + hi8;
        float gav = cg, bav = cb;
        for (int t = 0; t < 9; ++t) {
          int hh = h + t / 3 - 1, ww = w + t % 3 - 1;
          if (hh >= 0 && hh < H_ && ww >= 0 && ww < W_) {
            int labv = seg[(b * H_ + hh) * W_ + ww];
            const float* gp = Gtab + (size_t)(b * J_ + labv) * NTAB + t * C_ + c;
            gav += gp[0];
            bav += gp[2304];
          }
        }
        size_t oidx = ((size_t)(b * C_ + c) * H_ + h) * W_ + w;
        float xn = x[oidx] + noise[(b * W_ + w) * H_ + h] * nv;
        float nrm = (xn - mn) * rs;
        float gf = ga * gav + (1.f - ga) * gsp;
        float bfv = ba * bav + (1.f - ba) * bsp;
        out[oidx] = nrm * (1.f + gf) + bfv;
      }
    }
  }
}

extern "C" void kernel_launch(void* const* d_in, const int* in_sizes, int n_in,
                              void* d_out, int out_size, void* d_ws, size_t ws_size,
                              hipStream_t stream) {
  (void)in_sizes; (void)n_in; (void)out_size; (void)ws_size;
  const float* x         = (const float*)d_in[0];
  const int*   seg       = (const int*)d_in[1];
  const float* style     = (const float*)d_in[2];
  const float* noise     = (const float*)d_in[3];
  const float* noise_var = (const float*)d_in[4];
  const float* bgam      = (const float*)d_in[5];
  const float* bbet      = (const float*)d_in[6];
  const float* fc_w      = (const float*)d_in[7];
  const float* fc_b      = (const float*)d_in[8];
  const float* cgw       = (const float*)d_in[9];
  const float* cgbv      = (const float*)d_in[10];
  const float* cbw       = (const float*)d_in[11];
  const float* cbbv      = (const float*)d_in[12];
  const float* shw       = (const float*)d_in[13];
  const float* shb       = (const float*)d_in[14];
  const float* sgw       = (const float*)d_in[15];
  const float* sgbv      = (const float*)d_in[16];
  const float* sbw       = (const float*)d_in[17];
  const float* sbbv      = (const float*)d_in[18];
  float* out = (float*)d_out;

  char* ws = (char*)d_ws;
  auto take = [&](size_t bytes) -> char* {
    char* p = ws;
    ws += (bytes + 255) & ~(size_t)255;
    return p;
  };
  float*          mean  = (float*)take((size_t)B_ * C_ * 4);
  float*          rstd  = (float*)take((size_t)B_ * C_ * 4);
  unsigned short* mu_bf = (unsigned short*)take((size_t)BJPAD * S_ * 2);
  unsigned short* WtT   = (unsigned short*)take((size_t)NTAB * S_ * 2);
  float*          Gtab  = (float*)take((size_t)BJ_ * NTAB * 4);
  unsigned short* WspT  = (unsigned short*)take((size_t)2 * C_ * KSP * 2);
  float*          Tsh   = (float*)take((size_t)J_ * 9 * NH_ * 4);
  unsigned short* actv  = (unsigned short*)take((size_t)B_ * H_ * W_ * NH_ * 2);

  k_stats<<<dim3(B_ * C_), dim3(256), 0, stream>>>(x, noise, noise_var, mean, rstd);
  k_mu<<<dim3(BJPAD), dim3(256), 0, stream>>>(style, fc_w, fc_b, mu_bf);
  {
    size_t tot = (size_t)NTAB * S_;
    k_tr_gb<<<dim3((unsigned)((tot + 255) / 256)), dim3(256), 0, stream>>>(cgw, cbw, WtT);
  }
  {
    size_t tot = (size_t)2 * C_ * KSP;
    k_tr_sp<<<dim3((unsigned)((tot + 255) / 256)), dim3(256), 0, stream>>>(sgw, sbw, WspT);
  }
  {
    size_t tot = (size_t)J_ * 9 * NH_;
    k_tr_sh<<<dim3((unsigned)((tot + 255) / 256)), dim3(256), 0, stream>>>(shw, Tsh);
  }
  k_gemm_tab<<<dim3(BJPAD / 16, NTAB / 16 / 4), dim3(128), 0, stream>>>(mu_bf, WtT, Gtab);
  k_actv<<<dim3(B_ * H_), dim3(256), 0, stream>>>(seg, Tsh, shb, actv);
  k_main<<<dim3(B_ * H_ * 2, 4), dim3(128), 0, stream>>>(
      x, seg, noise, noise_var, bgam, bbet, cgbv, cbbv, sgbv, sbbv,
      mean, rstd, Gtab, actv, WspT, out);
}
